// SelfAttention_2491081032017
// MI455X (gfx1250) — matrix operand order repaired and checked by reading, not compiled, not run on hardware
//
#include <hip/hip_runtime.h>

// ---------------------------------------------------------------------------
// Self-attention for B=4, S=4096, D=512 (fp32 in/out) on gfx1250.
// All matmuls via v_wmma_f32_16x16x32_bf16 (wave32 WMMA), flash-attention
// structure. V-fragment hoisting + global_prefetch_b8 for latency hiding,
// double-buffered LDS score exchange (one barrier per key chunk).
// ---------------------------------------------------------------------------

#define B_DIM 4
#define S_DIM 4096
#define D_DIM 512
#define NE    (B_DIM * S_DIM * D_DIM)   // 8,388,608 elements per [B,S,D] tensor

typedef __attribute__((ext_vector_type(16))) __bf16 v16bf;
typedef __attribute__((ext_vector_type(8)))  __bf16 bf16x8;
typedef __attribute__((ext_vector_type(8)))  float  v8f;

static __device__ __forceinline__ __bf16 f2bf(float f) {
    unsigned u = __builtin_bit_cast(unsigned, f);
    unsigned r = (u + 0x7FFFu + ((u >> 16) & 1u)) >> 16;   // round-to-nearest-even
    return __builtin_bit_cast(__bf16, (unsigned short)r);
}

static __device__ __forceinline__ v16bf mk16(bf16x8 lo, bf16x8 hi) {
    return __builtin_shufflevector(lo, hi, 0,1,2,3,4,5,6,7,8,9,10,11,12,13,14,15);
}

static __device__ __forceinline__ v8f wmma_bf16(v16bf a, v16bf b, v8f c) {
    // D = A(16x32 bf16) x B(32x16 bf16) + C(16x16 f32)
    return __builtin_amdgcn_wmma_f32_16x16x32_bf16(false, a, false, b, (short)0, c,
                                                   false, false);
}

// A-fragment (16x32 bf16, row-major source, leading dim ld):
// lane holds row = lane&15; elems 0..7  <- K = half*8 + j
//                           elems 8..15 <- K = 16 + half*8 + j
static __device__ __forceinline__ v16bf load_afrag(const __bf16* base, int ld,
                                                   int row, int half) {
    const __bf16* p = base + (long)row * ld;
    bf16x8 lo = *(const bf16x8*)(p + half * 8);
    bf16x8 hi = *(const bf16x8*)(p + 16 + half * 8);
    return mk16(lo, hi);
}

// B-fragment (32x16 bf16) where B[k][n] is stored contiguously along k for
// fixed n (i.e. source is row-major [n][k] with leading dim ld):
// lane holds col n = lane&15; elems 0..7  <- K = half*8 + j
//                             elems 8..15 <- K = 16 + half*8 + j
static __device__ __forceinline__ v16bf load_bfrag(const __bf16* base, int ld,
                                                   int n, int half) {
    const __bf16* p = base + (long)n * ld;
    bf16x8 lo = *(const bf16x8*)(p + half * 8);
    bf16x8 hi = *(const bf16x8*)(p + 16 + half * 8);
    return mk16(lo, hi);
}

// ---------------------------------------------------------------------------
// Kernel 1: x (fp32) -> xb (bf16)
// ---------------------------------------------------------------------------
__global__ void cvt_x_kernel(const float* __restrict__ x, __bf16* __restrict__ xb) {
    int i = (blockIdx.x * 256 + threadIdx.x) * 4;
#pragma unroll
    for (int j = 0; j < 4; ++j) xb[i + j] = f2bf(x[i + j]);
}

// ---------------------------------------------------------------------------
// Kernel 2: transpose+convert weights: WT[e*D + d] = bf16(W[d*D + e])
// ---------------------------------------------------------------------------
__global__ void prep_w_kernel(const float* __restrict__ Wq, const float* __restrict__ Wk,
                              const float* __restrict__ Wv,
                              __bf16* __restrict__ WqT, __bf16* __restrict__ WkT,
                              __bf16* __restrict__ WvT) {
    int i = blockIdx.x * 256 + threadIdx.x;   // i = d*D + e
    int d = i / D_DIM, e = i % D_DIM;
    int t = e * D_DIM + d;
    WqT[t] = f2bf(Wq[i]);
    WkT[t] = f2bf(Wk[i]);
    WvT[t] = f2bf(Wv[i]);
}

// ---------------------------------------------------------------------------
// Kernel 3: QKV projection.  One wave per 16x16 output tile.
//   Qb[s][e] = (x@Wq + bq)[s][e] * (1/sqrt(D))   (bf16)
//   Kb[s][e] = (x@Wk + bk)[s][e]                 (bf16)
//   Vt[e][s] = (x@Wv + bv)[s][e]                 (bf16, transposed)
// ---------------------------------------------------------------------------
__global__ void qkv_proj_kernel(const __bf16* __restrict__ xb,
                                const __bf16* __restrict__ WqT,
                                const __bf16* __restrict__ WkT,
                                const __bf16* __restrict__ WvT,
                                const float* __restrict__ bq,
                                const float* __restrict__ bk,
                                const float* __restrict__ bv,
                                __bf16* __restrict__ Qb,
                                __bf16* __restrict__ Kb,
                                __bf16* __restrict__ Vt) {
    const int wave = blockIdx.x * 8 + (threadIdx.x >> 5);
    const int lane = threadIdx.x & 31;
    const int row = lane & 15, half = lane >> 4;

    const int eTiles = D_DIM / 16;                 // 32
    const int rowTile = wave / eTiles;             // [0, B*S/16)
    const int eTile = wave % eTiles;
    const int b = rowTile >> 8;                    // S/16 = 256
    const int s0 = (rowTile & 255) * 16;
    const int e0 = eTile * 16;

    const __bf16* xrow = xb + (long)(b * S_DIM + s0) * D_DIM;

    v8f accQ = (v8f)0.0f, accK = (v8f)0.0f, accV = (v8f)0.0f;
#pragma unroll 4
    for (int kc = 0; kc < D_DIM / 32; ++kc) {
        const int k0 = kc * 32;
        v16bf a  = load_afrag(xrow + k0, D_DIM, row, half);
        v16bf bQ = load_bfrag(WqT + (long)e0 * D_DIM + k0, D_DIM, row, half);
        v16bf bK = load_bfrag(WkT + (long)e0 * D_DIM + k0, D_DIM, row, half);
        v16bf bV = load_bfrag(WvT + (long)e0 * D_DIM + k0, D_DIM, row, half);
        accQ = wmma_bf16(a, bQ, accQ);
        accK = wmma_bf16(a, bK, accK);
        accV = wmma_bf16(a, bV, accV);
    }

    const float inv_sqrt_d = 0.044194173824159216f;   // 1/sqrt(512)
    const float bqv = bq[e0 + row];
    const float bkv = bk[e0 + row];
    const float bvv = bv[e0 + row];
#pragma unroll
    for (int r = 0; r < 8; ++r) {
        const int s = s0 + half * 8 + r;                      // C-layout row M
        const long qk_idx = (long)(b * S_DIM + s) * D_DIM + e0 + row;
        Qb[qk_idx] = f2bf((accQ[r] + bqv) * inv_sqrt_d);
        Kb[qk_idx] = f2bf(accK[r] + bkv);
        Vt[(long)(b * D_DIM + e0 + row) * S_DIM + s] = f2bf(accV[r] + bvv);
    }
}

// ---------------------------------------------------------------------------
// Kernel 4: flash attention.  One block (4 waves) per (b, 16-query tile).
// Wave w owns feature slice [w*128, w*128+128) of D for both the Q.K^T
// partial sums and the output accumulator.
// ---------------------------------------------------------------------------
__global__ void flash_attn_kernel(const __bf16* __restrict__ Qb,
                                  const __bf16* __restrict__ Kb,
                                  const __bf16* __restrict__ Vt,
                                  float* __restrict__ out) {
    __shared__ float sPart[2][4][16][32];  // double-buffered partial score tiles
    __shared__ float sAlphaW[4][16];       // per-wave row rescale factors
    __shared__ float sLW[4][16];           // per-wave row softmax denominators

    const int w = threadIdx.x >> 5;
    const int lane = threadIdx.x & 31;
    const int row = lane & 15, half = lane >> 4;

    const int b = blockIdx.x >> 8;         // S/16 = 256 query tiles per batch
    const int q0 = (blockIdx.x & 255) * 16;

    const __bf16* Kbase = Kb + (long)(b * S_DIM) * D_DIM;
    const __bf16* Vbase = Vt + (long)(b * D_DIM) * S_DIM;

    // Preload this wave's Q A-fragments (invariant across key iterations).
    v16bf qf[4];
#pragma unroll
    for (int c = 0; c < 4; ++c) {
        const int d0 = w * 128 + c * 32;
        qf[c] = load_afrag(Qb + (long)(b * S_DIM + q0) * D_DIM + d0, D_DIM, row, half);
    }

    v8f o[8];
#pragma unroll
    for (int t = 0; t < 8; ++t) o[t] = (v8f)0.0f;
    float m_run = -1.0e30f, l_run = 0.0f;

    for (int kc = 0; kc < S_DIM / 32; ++kc) {
        const int kbase = kc * 32;
        const int buf = kc & 1;

        // ---- partial scores: Q(16 x 128slice) . K^T  -> two 16x16 tiles ----
        v8f sc0 = (v8f)0.0f, sc1 = (v8f)0.0f;
#pragma unroll
        for (int c = 0; c < 4; ++c) {
            const int d0 = w * 128 + c * 32;
            v16bf bk0 = load_bfrag(Kbase + (long)kbase * D_DIM + d0, D_DIM, row, half);
            v16bf bk1 = load_bfrag(Kbase + (long)(kbase + 16) * D_DIM + d0,
                                   D_DIM, row, half);
            sc0 = wmma_bf16(qf[c], bk0, sc0);
            sc1 = wmma_bf16(qf[c], bk1, sc1);
        }

        // ---- hoist this chunk's V fragments: their L2 latency overlaps the
        // ---- LDS reduction + softmax VALU work below.
        v16bf vf[8];
#pragma unroll
        for (int t = 0; t < 8; ++t) {
            const int e0 = w * 128 + t * 16;
            vf[t] = load_bfrag(Vbase + (long)e0 * S_DIM + kbase, S_DIM, row, half);
        }

#pragma unroll
        for (int r = 0; r < 8; ++r) {
            const int M = half * 8 + r;
            sPart[buf][w][M][row]      = sc0[r];
            sPart[buf][w][M][16 + row] = sc1[r];
        }
        __syncthreads();

        // ---- online softmax (each wave computes full row stats) ----
        // This lane handles the 16 columns of its A-fragment layout; the
        // complementary 16 columns of the same row live in lane^16.
        float sv[16];
        float mx = -1.0e30f;
#pragma unroll
        for (int j = 0; j < 8; ++j) {
            const int c0 = half * 8 + j;
            const int c1 = 16 + half * 8 + j;
            float a0 = sPart[buf][0][row][c0] + sPart[buf][1][row][c0] +
                       sPart[buf][2][row][c0] + sPart[buf][3][row][c0];
            float a1 = sPart[buf][0][row][c1] + sPart[buf][1][row][c1] +
                       sPart[buf][2][row][c1] + sPart[buf][3][row][c1];
            sv[j] = a0;
            sv[8 + j] = a1;
            mx = fmaxf(mx, fmaxf(a0, a1));
        }
        mx = fmaxf(mx, __shfl_xor(mx, 16, 32));          // full row max
        const float m_new = fmaxf(m_run, mx);
        const float alpha = __expf(m_run - m_new);

        v16bf pf;
        float psum = 0.0f;
#pragma unroll
        for (int j = 0; j < 16; ++j) {
            float p = __expf(sv[j] - m_new);
            psum += p;
            pf[j] = f2bf(p);
        }
        psum += __shfl_xor(psum, 16, 32);                // full row sum
        l_run = l_run * alpha + psum;
        m_run = m_new;

        if (lane < 16) sAlphaW[w][row] = alpha;          // broadcast within wave
        float am[8];
#pragma unroll
        for (int r = 0; r < 8; ++r) am[r] = sAlphaW[w][half * 8 + r];

        // ---- rescale O and accumulate P x V ----
#pragma unroll
        for (int t = 0; t < 8; ++t) {
#pragma unroll
            for (int r = 0; r < 8; ++r) o[t][r] *= am[r];
            o[t] = wmma_bf16(pf, vf[t], o[t]);
        }

        // ---- warm next chunk's K and V rows (global_prefetch_b8) ----
        if (kc + 1 < S_DIM / 32) {
            const int kn = kbase + 32;
            __builtin_prefetch(Kbase + (long)(kn + row) * D_DIM + w * 128, 0, 3);
            __builtin_prefetch(Kbase + (long)(kn + 16 + row) * D_DIM + w * 128, 0, 3);
            __builtin_prefetch(Vbase + (long)(w * 128 + row) * S_DIM + kn, 0, 3);
            __builtin_prefetch(Vbase + (long)(w * 128 + 64 + row) * S_DIM + kn, 0, 3);
        }
        // No trailing barrier: next iteration writes the other sPart buffer,
        // and reuse of this buffer is fenced by the next iteration's barrier.
    }

    // ---- final normalization and store (fp32) ----
    if (lane < 16) sLW[w][row] = l_run;
    float lm[8];
#pragma unroll
    for (int r = 0; r < 8; ++r) lm[r] = 1.0f / sLW[w][half * 8 + r];

#pragma unroll
    for (int t = 0; t < 8; ++t) {
        const int e0 = w * 128 + t * 16;
#pragma unroll
        for (int r = 0; r < 8; ++r) {
            const int s = q0 + half * 8 + r;
            out[(long)(b * S_DIM + s) * D_DIM + e0 + row] = o[t][r] * lm[r];
        }
    }
}

// ---------------------------------------------------------------------------
extern "C" void kernel_launch(void* const* d_in, const int* in_sizes, int n_in,
                              void* d_out, int out_size, void* d_ws, size_t ws_size,
                              hipStream_t stream) {
    const float* x  = (const float*)d_in[0];
    const float* Wq = (const float*)d_in[1];
    const float* bq = (const float*)d_in[2];
    const float* Wk = (const float*)d_in[3];
    const float* bk = (const float*)d_in[4];
    const float* Wv = (const float*)d_in[5];
    const float* bv = (const float*)d_in[6];
    float* out = (float*)d_out;

    __bf16* xb  = (__bf16*)d_ws;
    __bf16* Qb  = xb + NE;
    __bf16* Kb  = Qb + NE;
    __bf16* Vt  = Kb + NE;
    __bf16* WqT = Vt + NE;
    __bf16* WkT = WqT + D_DIM * D_DIM;
    __bf16* WvT = WkT + D_DIM * D_DIM;

    // 1) x -> bf16
    cvt_x_kernel<<<NE / (256 * 4), 256, 0, stream>>>(x, xb);
    // 2) transpose+convert weights
    prep_w_kernel<<<(D_DIM * D_DIM) / 256, 256, 0, stream>>>(Wq, Wk, Wv, WqT, WkT, WvT);
    // 3) QKV projection: (B*S/16)*(D/16) = 32768 waves, 8 waves/block
    qkv_proj_kernel<<<4096, 256, 0, stream>>>(xb, WqT, WkT, WvT, bq, bk, bv, Qb, Kb, Vt);
    // 4) flash attention: one block per (b, 16-query tile)
    flash_attn_kernel<<<B_DIM * (S_DIM / 16), 128, 0, stream>>>(Qb, Kb, Vt, out);
}
